// Net_46608985096658
// MI455X (gfx1250) — matrix operand order repaired and checked by reading, not compiled, not run on hardware
//
#include <hip/hip_runtime.h>
#include <math.h>

#define N_NODES 100000
#define N_EDGES 1600000
#define G_GRAPHS 64
#define DIM 128
#define OUT_DIM 10
#define LAYERS 3
#define BN_EPS 1e-5f

typedef __attribute__((ext_vector_type(16))) __bf16 v16bf;
typedef __attribute__((ext_vector_type(8)))  float  v8f;

// ---------------- utility kernels ----------------
__global__ void k_zero(float* __restrict__ p, int n) {
    int i = blockIdx.x * blockDim.x + threadIdx.x;
    if (i < n) p[i] = 0.f;
}

__global__ void k_copy(const float4* __restrict__ in, float4* __restrict__ out, int n4) {
    int i = blockIdx.x * blockDim.x + threadIdx.x;
    if (i < n4) out[i] = in[i];
}

// scatter-add: out[dst[e]] += h[src[e]]  (out pre-initialized with h -> gives h + agg)
// one wave per edge, lane handles 4 channels via float4
__global__ void k_scatter(const float* __restrict__ h, const int* __restrict__ src,
                          const int* __restrict__ dst, float* __restrict__ out) {
    int wid  = (blockIdx.x * blockDim.x + threadIdx.x) >> 5;
    int lane = threadIdx.x & 31;
    if (wid >= N_EDGES) return;
    int s = src[wid], d = dst[wid];
    float4 v = ((const float4*)(h + (size_t)s * DIM))[lane];
    float* o = out + (size_t)d * DIM + lane * 4;
    atomicAdd(o + 0, v.x); atomicAdd(o + 1, v.y);
    atomicAdd(o + 2, v.z); atomicAdd(o + 3, v.w);
}

// pack one 128x128 f32 weight (k-major, W[k*128+n]) into bf16 WMMA B-fragment order.
// chunk index c = (ntile*4 + kblock)*32 + lane ; element i within chunk:
//   n = ntile*16 + (lane&15), K = kblock*32 + (lane>>4)*16 + i    (contiguous k order, same as A)
__global__ void k_pack(const float* __restrict__ W, __bf16* __restrict__ dst) {
    int t = blockIdx.x * blockDim.x + threadIdx.x;   // 16384 threads per matrix
    if (t >= DIM * DIM) return;
    int i    = t & 15;
    int lane = (t >> 4) & 31;
    int kb   = (t >> 9) & 3;
    int nt   = t >> 11;
    int hlf  = lane >> 4;
    int n = nt * 16 + (lane & 15);
    int k = kb * 32 + hlf * 16 + i;
    dst[t] = (__bf16)W[k * DIM + n];
}

// Y[N,128] = relu(X[N,128] @ W + bias), W pre-packed bf16 fragments, f32 accumulate.
// block = 256 threads = 8 waves; block -> 16 rows, wave w -> cols 16w..16w+15.
__global__ __launch_bounds__(256)
void k_gemm_relu(const float* __restrict__ X, const v16bf* __restrict__ Wp,
                 const float* __restrict__ bias, float* __restrict__ Y) {
    int lane = threadIdx.x & 31;
    int wave = threadIdx.x >> 5;          // column tile 0..7
    int hlf  = lane >> 4;
    int m    = lane & 15;
    int rowbase = blockIdx.x * 16;

    const float4* Xr = (const float4*)(X + (size_t)(rowbase + m) * DIM);
    v8f acc = {};
#pragma unroll
    for (int kb = 0; kb < 4; ++kb) {
        // A fragment (16x32 bf16): element i <-> K = kb*32 + hlf*16 + i
        int q0 = (kb * 32 + hlf * 16) >> 2;           // float4 index of first run
        float4 a0 = Xr[q0], a1 = Xr[q0 + 1];          // K = base .. base+7
        float4 a2 = Xr[q0 + 2], a3 = Xr[q0 + 3];      // K = base+8 .. base+15
        float fa[16];
        *(float4*)(fa + 0)  = a0; *(float4*)(fa + 4)  = a1;
        *(float4*)(fa + 8)  = a2; *(float4*)(fa + 12) = a3;
        v16bf A;
#pragma unroll
        for (int i = 0; i < 16; ++i) A[i] = (__bf16)fa[i];

        v16bf B = Wp[(wave * 4 + kb) * 32 + lane];    // contiguous 32B fragment

        acc = __builtin_amdgcn_wmma_f32_16x16x32_bf16(
            /*neg_a=*/false, A, /*neg_b=*/false, B,
            /*c_mod=*/(short)0, acc, /*reuse_a=*/false, /*reuse_b=*/false);
    }
    // C layout: VGPR r -> M = hlf*8 + r, N = lane&15
    int col = wave * 16 + m;
    float bb = bias[col];
#pragma unroll
    for (int r = 0; r < 8; ++r) {
        int row = rowbase + hlf * 8 + r;
        float v = acc[r] + bb;
        Y[(size_t)row * DIM + col] = v > 0.f ? v : 0.f;
    }
}

// per-channel mean/var over all N nodes (one block per channel)
__global__ void k_bnstats(const float* __restrict__ Y, float* __restrict__ mu,
                          float* __restrict__ var) {
    int c = blockIdx.x;
    float s = 0.f, ss = 0.f;
    for (int i = threadIdx.x; i < N_NODES; i += blockDim.x) {
        float v = Y[(size_t)i * DIM + c];
        s += v; ss += v * v;
    }
    __shared__ float sh[256], sh2[256];
    sh[threadIdx.x] = s; sh2[threadIdx.x] = ss;
    __syncthreads();
    for (int o = 128; o > 0; o >>= 1) {
        if (threadIdx.x < o) { sh[threadIdx.x] += sh[threadIdx.x + o];
                               sh2[threadIdx.x] += sh2[threadIdx.x + o]; }
        __syncthreads();
    }
    if (threadIdx.x == 0) {
        float mn = sh[0] / (float)N_NODES;
        mu[c] = mn;
        var[c] = sh2[0] / (float)N_NODES - mn * mn;
    }
}

__global__ void k_bnapply(const float* __restrict__ Y, const float* __restrict__ mu,
                          const float* __restrict__ var, const float* __restrict__ g,
                          const float* __restrict__ bt, float* __restrict__ O) {
    int i = blockIdx.x * blockDim.x + threadIdx.x;
    if (i >= N_NODES * DIM) return;
    int c = i & (DIM - 1);
    O[i] = (Y[i] - mu[c]) * rsqrtf(var[c] + BN_EPS) * g[c] + bt[c];
}

__global__ void k_counts(const int* __restrict__ batch, float* __restrict__ counts) {
    int i = blockIdx.x * blockDim.x + threadIdx.x;
    if (i < N_NODES) atomicAdd(&counts[batch[i]], 1.0f);
}

// segment-sum pool: one wave per node
__global__ void k_pool(const float* __restrict__ h, const int* __restrict__ batch,
                       float* __restrict__ pool) {
    int wid  = (blockIdx.x * blockDim.x + threadIdx.x) >> 5;
    int lane = threadIdx.x & 31;
    if (wid >= N_NODES) return;
    int b = batch[wid];
    float4 v = ((const float4*)(h + (size_t)wid * DIM))[lane];
    float* o = pool + b * DIM + lane * 4;
    atomicAdd(o + 0, v.x); atomicAdd(o + 1, v.y);
    atomicAdd(o + 2, v.z); atomicAdd(o + 3, v.w);
}

__global__ void k_div(const float* __restrict__ poolA, const float* __restrict__ poolB,
                      const float* __restrict__ counts, float* __restrict__ meanA,
                      float* __restrict__ poolX) {
    int i = blockIdx.x * blockDim.x + threadIdx.x;
    if (i >= G_GRAPHS * DIM) return;
    float c = counts[i >> 7];
    c = c < 1.f ? 1.f : c;
    meanA[i] = poolA[i] / c;
    poolX[i] = poolB[i] / c;
}

// bilinear discriminator logits: out[g] = px[g]ᵀ W px[g] + b ; out[G+g] with shuffled rhs
__global__ void k_disc(const float* __restrict__ px, const float* __restrict__ W,
                       const float* __restrict__ db, float* __restrict__ out) {
    int g = blockIdx.x;
    int sg = (g == G_GRAPHS / 2) ? (G_GRAPHS - 1 - (G_GRAPHS / 2 + 1)) : (G_GRAPHS - 1 - g);
    const float* a  = px + g * DIM;
    const float* c2 = px + sg * DIM;
    float s1 = 0.f, s2 = 0.f;
    for (int d = threadIdx.x; d < DIM; d += blockDim.x) {
        float t1 = 0.f, t2 = 0.f;
        const float* Wd = W + d * DIM;
        for (int e = 0; e < DIM; ++e) { float w = Wd[e]; t1 += w * a[e]; t2 += w * c2[e]; }
        s1 += a[d] * t1; s2 += a[d] * t2;
    }
    __shared__ float r1[128], r2[128];
    r1[threadIdx.x] = s1; r2[threadIdx.x] = s2;
    __syncthreads();
    for (int o = 64; o > 0; o >>= 1) {
        if (threadIdx.x < o) { r1[threadIdx.x] += r1[threadIdx.x + o];
                               r2[threadIdx.x] += r2[threadIdx.x + o]; }
        __syncthreads();
    }
    if (threadIdx.x == 0) {
        float b = db[0];
        out[g] = r1[0] + b;
        out[G_GRAPHS + g] = r2[0] + b;
    }
}

// classification head: cat(meanA, poolX) -> lin1 -> relu -> lin2 -> log_softmax
__global__ void k_head(const float* __restrict__ meanA, const float* __restrict__ px,
                       const float* __restrict__ w1, const float* __restrict__ b1,
                       const float* __restrict__ w2, const float* __restrict__ b2,
                       float* __restrict__ out) {
    int g = blockIdx.x;
    int j = threadIdx.x;                       // 128 threads
    __shared__ float z1[DIM];
    __shared__ float z2[OUT_DIM];
    const float* za = meanA + g * DIM;
    const float* zb = px + g * DIM;
    float acc = b1[j];
    for (int k = 0; k < DIM; ++k) acc += za[k] * w1[k * DIM + j];
    for (int k = 0; k < DIM; ++k) acc += zb[k] * w1[(DIM + k) * DIM + j];
    z1[j] = acc > 0.f ? acc : 0.f;
    __syncthreads();
    if (j < OUT_DIM) {
        float a = b2[j];
        for (int k = 0; k < DIM; ++k) a += z1[k] * w2[k * OUT_DIM + j];
        z2[j] = a;
    }
    __syncthreads();
    if (j == 0) {
        float m = z2[0];
        for (int o = 1; o < OUT_DIM; ++o) m = fmaxf(m, z2[o]);
        float s = 0.f;
        for (int o = 0; o < OUT_DIM; ++o) s += expf(z2[o] - m);
        float lse = m + logf(s);
        for (int o = 0; o < OUT_DIM; ++o) out[g * OUT_DIM + o] = z2[o] - lse;
    }
}

// ---------------- host-side launcher ----------------
extern "C" void kernel_launch(void* const* d_in, const int* in_sizes, int n_in,
                              void* d_out, int out_size, void* d_ws, size_t ws_size,
                              hipStream_t stream) {
    (void)in_sizes; (void)n_in; (void)out_size; (void)ws_size;
    const float* x      = (const float*)d_in[0];
    const int*   ei     = (const int*)d_in[1];
    const int*   batch  = (const int*)d_in[2];
    const float* W1a    = (const float*)d_in[3];
    const float* b1a    = (const float*)d_in[4];
    const float* W2a    = (const float*)d_in[5];
    const float* b2a    = (const float*)d_in[6];
    const float* ga     = (const float*)d_in[7];
    const float* bta    = (const float*)d_in[8];
    const float* W1b    = (const float*)d_in[9];
    const float* b1b    = (const float*)d_in[10];
    const float* W2b    = (const float*)d_in[11];
    const float* b2b    = (const float*)d_in[12];
    const float* gb     = (const float*)d_in[13];
    const float* btb    = (const float*)d_in[14];
    const float* lin1_w = (const float*)d_in[15];
    const float* lin1_b = (const float*)d_in[16];
    const float* lin2_w = (const float*)d_in[17];
    const float* lin2_b = (const float*)d_in[18];
    const float* disc_w = (const float*)d_in[19];
    const float* disc_b = (const float*)d_in[20];
    float* out = (float*)d_out;

    const int* src  = ei;
    const int* dstp = ei + N_EDGES;

    // workspace layout (256B aligned blocks)
    char* ws = (char*)d_ws;
    size_t off = 0;
    auto alloc = [&](size_t bytes) -> char* {
        char* p = ws + off;
        off += (bytes + 255) & ~(size_t)255;
        return p;
    };
    float*  bufA   = (float*)alloc(sizeof(float) * (size_t)N_NODES * DIM);
    float*  bufB   = (float*)alloc(sizeof(float) * (size_t)N_NODES * DIM);
    __bf16* packed = (__bf16*)alloc(sizeof(__bf16) * 12 * DIM * DIM);
    float*  mu     = (float*)alloc(sizeof(float) * DIM);
    float*  var    = (float*)alloc(sizeof(float) * DIM);
    float*  poolA  = (float*)alloc(sizeof(float) * G_GRAPHS * DIM);
    float*  poolB  = (float*)alloc(sizeof(float) * G_GRAPHS * DIM);
    float*  counts = (float*)alloc(sizeof(float) * G_GRAPHS);
    float*  meanA  = (float*)alloc(sizeof(float) * G_GRAPHS * DIM);
    float*  poolX  = (float*)alloc(sizeof(float) * G_GRAPHS * DIM);

    // pack all 12 weight matrices: order [W1a(3), W2a(3), W1b(3), W2b(3)]
    const float* wsrc[4] = {W1a, W2a, W1b, W2b};
    for (int s = 0; s < 4; ++s)
        for (int l = 0; l < LAYERS; ++l)
            k_pack<<<(DIM * DIM) / 256, 256, 0, stream>>>(
                wsrc[s] + (size_t)l * DIM * DIM,
                packed + (size_t)(s * LAYERS + l) * DIM * DIM);

    // poolA, poolB, counts are contiguous in ws (each block multiple of 256B)
    k_zero<<<(2 * G_GRAPHS * DIM + G_GRAPHS + 255) / 256, 256, 0, stream>>>(
        poolA, 2 * G_GRAPHS * DIM + G_GRAPHS);
    k_counts<<<(N_NODES + 255) / 256, 256, 0, stream>>>(batch, counts);

    const int n4            = N_NODES * DIM / 4;
    const int copyBlocks    = (n4 + 255) / 256;
    const int scatterBlocks = (N_EDGES * 32 + 255) / 256;   // wave per edge
    const int poolBlocks    = (N_NODES * 32 + 255) / 256;   // wave per node
    const int gemmBlocks    = N_NODES / 16;                 // 6250, exact
    const int elemBlocks    = (N_NODES * DIM + 255) / 256;

    for (int br = 0; br < 2; ++br) {
        const float* cur = x;
        const float* bb1 = br ? b1b : b1a;
        const float* bb2 = br ? b2b : b2a;
        const float* gg  = br ? gb : ga;
        const float* bt  = br ? btb : bta;
        int w1base = br ? 6 : 0;
        int w2base = br ? 9 : 3;
        for (int l = 0; l < LAYERS; ++l) {
            // tmp = cur + agg  (copy then edge scatter-add)
            k_copy<<<copyBlocks, 256, 0, stream>>>((const float4*)cur, (float4*)bufB, n4);
            k_scatter<<<scatterBlocks, 256, 0, stream>>>(cur, src, dstp, bufB);
            // y1 = relu(tmp @ W1 + b1)
            k_gemm_relu<<<gemmBlocks, 256, 0, stream>>>(
                bufB, (const v16bf*)(packed + (size_t)(w1base + l) * DIM * DIM),
                bb1 + l * DIM, bufA);
            // y2 = relu(y1 @ W2 + b2)
            k_gemm_relu<<<gemmBlocks, 256, 0, stream>>>(
                bufA, (const v16bf*)(packed + (size_t)(w2base + l) * DIM * DIM),
                bb2 + l * DIM, bufB);
            // batch norm
            k_bnstats<<<DIM, 256, 0, stream>>>(bufB, mu, var);
            k_bnapply<<<elemBlocks, 256, 0, stream>>>(bufB, mu, var,
                                                      gg + l * DIM, bt + l * DIM, bufA);
            cur = bufA;
        }
        k_pool<<<poolBlocks, 256, 0, stream>>>(cur, batch, br ? poolB : poolA);
    }

    k_div<<<(G_GRAPHS * DIM + 255) / 256, 256, 0, stream>>>(poolA, poolB, counts, meanA, poolX);
    k_disc<<<G_GRAPHS, 128, 0, stream>>>(poolX, disc_w, disc_b, out + G_GRAPHS * OUT_DIM);
    k_head<<<G_GRAPHS, 128, 0, stream>>>(meanA, poolX, lin1_w, lin1_b, lin2_w, lin2_b, out);
}
